// PAIR_MODULE_38723425140804
// MI455X (gfx1250) — matrix operand order repaired and checked by reading, not compiled, not run on hardware
//
#include <hip/hip_runtime.h>
#include <hip/hip_bf16.h>
#include <math.h>
#include <stdint.h>

typedef __attribute__((ext_vector_type(16))) _Float16 v16h;
typedef __attribute__((ext_vector_type(8)))  _Float16 v8h;
typedef __attribute__((ext_vector_type(4)))  _Float16 v4h;
typedef __attribute__((ext_vector_type(8)))  float    v8f;

#define BB    1024
#define RR    10
#define LLn   512            // L (GEMM N)
#define KK    1024           // 2L (GEMM K)
#define MM    (BB*RR)        // 10240 rows
#define MTILE 64             // rows per workgroup (4 waves x 16)
#define NPASS 128            // cols per accumulation pass
#define NT    8              // 16-col WMMA tiles per pass
#define KC    32             // K chunk per LDS stage (WMMA K)
#define APAD  40             // padded K-halves per LDS row (bank-spread, 16B aligned)
#define EPSF  1e-10f

// ---------- kernel 0a: f32 -> f16 (same layout), vectorized x4 ----------
__global__ void cvt_x(const float* __restrict__ in, _Float16* __restrict__ out, int n4) {
  int i = blockIdx.x * blockDim.x + threadIdx.x;
  if (i >= n4) return;
  float4 v = ((const float4*)in)[i];
  v4h h = { (_Float16)v.x, (_Float16)v.y, (_Float16)v.z, (_Float16)v.w };
  ((v4h*)out)[i] = h;
}

// ---------- kernel 0b: weights f32 [K=1024][N=512] -> f16 transposed [N=512][K=1024] ----------
__global__ void cvt_wt(const float* __restrict__ W, _Float16* __restrict__ Wt) {
  int idx = blockIdx.x * blockDim.x + threadIdx.x;
  if (idx >= LLn * KK) return;
  int n = idx >> 10;             // /1024
  int k = idx & 1023;
  Wt[idx] = (_Float16)W[(size_t)k * LLn + n];
}

// ---------- kernel 1: fold MLP layer 2: u = W1 @ W2, c0 = b1.W2 + b2 ----------
__global__ void fold_mlp(const float* __restrict__ W1, const float* __restrict__ b1,
                         const float* __restrict__ W2, const float* __restrict__ b2,
                         float* __restrict__ u, float* __restrict__ c0) {
  int k = blockIdx.x * blockDim.x + threadIdx.x;
  if (k < KK) {
    const float* row = W1 + (size_t)k * LLn;
    float s = 0.f;
    for (int l = 0; l < LLn; ++l) s += row[l] * W2[l];
    u[k] = s;
  } else if (k == KK) {
    float s = 0.f;
    for (int l = 0; l < LLn; ++l) s += b1[l] * W2[l];
    *c0 = s + b2[0];
  }
}

// async 16B global -> LDS copy, tracked by ASYNCcnt (CDNA5)
__device__ __forceinline__ void async_cp16(const _Float16* g, _Float16* l) {
  uint32_t lds_off = (uint32_t)(uintptr_t)l;          // low 32 bits of flat = LDS offset
  uint64_t gaddr   = (uint64_t)(uintptr_t)g;
  asm volatile("global_load_async_to_lds_b128 %0, %1, off"
               :: "v"(lds_off), "v"(gaddr) : "memory");
}
__device__ __forceinline__ void wait_async0() {
  asm volatile("s_wait_asynccnt 0x0" ::: "memory");
}

// stage one K-chunk: A tile 64x32 halves, B tiles 128x32 halves (x2), all via async DMA
__device__ __forceinline__ void stage_tile(const _Float16* __restrict__ X16,
                                           const _Float16* __restrict__ Wyt,
                                           const _Float16* __restrict__ Wgt,
                                           _Float16* sA, _Float16* sBy, _Float16* sBg,
                                           int mBase, int p, int kc, int tid) {
  // A: 64 rows x 4 chunks(8 halves) = 256 chunks, 2 per thread
  #pragma unroll
  for (int i = 0; i < 2; ++i) {
    int c = tid + i * 128;
    int row = c >> 2;
    int ko  = (c & 3) * 8;
    async_cp16(X16 + (size_t)(mBase + row) * KK + kc * KC + ko, sA + row * APAD + ko);
  }
  // B: 128 n-rows x 4 chunks = 512 chunks per matrix, 4 per thread per matrix
  #pragma unroll
  for (int i = 0; i < 4; ++i) {
    int c = tid + i * 128;
    int n = c >> 2;
    int ko = (c & 3) * 8;
    size_t gi = (size_t)(p * NPASS + n) * KK + kc * KC + ko;
    int li = n * APAD + ko;
    async_cp16(Wyt + gi, sBy + li);
    async_cp16(Wgt + gi, sBg + li);
  }
}

// ---------- kernel 2: v[m] = sum_n tanh(XWy+by) * sigmoid(XWg+bg) * u[n] ----------
__launch_bounds__(128)
__global__ void gated_gemm(const _Float16* __restrict__ X16,   // [M][1024] f16
                           const _Float16* __restrict__ Wyt,   // [512][1024] f16 (N-major)
                           const _Float16* __restrict__ Wgt,
                           const float* __restrict__ by, const float* __restrict__ bg,
                           const float* __restrict__ u,  float* __restrict__ vout) {
  __shared__ alignas(16) _Float16 smA [2][MTILE * APAD];
  __shared__ alignas(16) _Float16 smBy[2][NPASS * APAD];
  __shared__ alignas(16) _Float16 smBg[2][NPASS * APAD];

  const int tid   = threadIdx.x;
  const int lane  = tid & 31;
  const int wave  = tid >> 5;          // 0..3 -> 16-row slice
  const int grp   = lane >> 4;         // lane half (ISA A/B K-split)
  const int l16   = lane & 15;
  const int mBase = blockIdx.x * MTILE;

  float vsum[8];
  #pragma unroll
  for (int r = 0; r < 8; ++r) vsum[r] = 0.f;

  for (int p = 0; p < (LLn / NPASS); ++p) {             // 4 N passes of 128 cols
    v8f accY[NT], accG[NT];
    #pragma unroll
    for (int t = 0; t < NT; ++t) {
      accY[t] = (v8f){0.f,0.f,0.f,0.f,0.f,0.f,0.f,0.f};
      accG[t] = accY[t];
    }
    __syncthreads();                                     // prior reads of buf0 done
    stage_tile(X16, Wyt, Wgt, smA[0], smBy[0], smBg[0], mBase, p, 0, tid);

    for (int kc = 0; kc < (KK / KC); ++kc) {             // 32 K chunks, double-buffered
      int cur = kc & 1;
      wait_async0();                                     // this wave's copies done
      __syncthreads();                                   // all waves' copies visible
      if (kc + 1 < (KK / KC))
        stage_tile(X16, Wyt, Wgt, smA[cur ^ 1], smBy[cur ^ 1], smBg[cur ^ 1],
                   mBase, p, kc + 1, tid);
      // A fragment: lanes 0-15 K=0..15 ; lanes 16-31 K=16..31
      const _Float16* pa = &smA[cur][(wave * 16 + l16) * APAD + grp * 16];
      v8h alo = *(const v8h*)pa;
      v8h ahi = *(const v8h*)(pa + 8);
      v16h afrag = __builtin_shufflevector(alo, ahi, 0,1,2,3,4,5,6,7,8,9,10,11,12,13,14,15);
      #pragma unroll
      for (int t = 0; t < NT; ++t) {
        const _Float16* pby = &smBy[cur][(t * 16 + l16) * APAD + grp * 16];
        v8h ylo = *(const v8h*)pby;
        v8h yhi = *(const v8h*)(pby + 8);
        v16h bfy = __builtin_shufflevector(ylo, yhi, 0,1,2,3,4,5,6,7,8,9,10,11,12,13,14,15);
        accY[t] = __builtin_amdgcn_wmma_f32_16x16x32_f16(
            false, afrag, false, bfy, (short)0, accY[t], false, false);
        const _Float16* pbg = &smBg[cur][(t * 16 + l16) * APAD + grp * 16];
        v8h glo = *(const v8h*)pbg;
        v8h ghi = *(const v8h*)(pbg + 8);
        v16h bfg = __builtin_shufflevector(glo, ghi, 0,1,2,3,4,5,6,7,8,9,10,11,12,13,14,15);
        accG[t] = __builtin_amdgcn_wmma_f32_16x16x32_f16(
            false, afrag, false, bfg, (short)0, accG[t], false, false);
      }
    }
    // fused epilogue: gate + dot with u (C layout: VGPR r -> M=r / r+8 by lane half, N=l16)
    #pragma unroll
    for (int t = 0; t < NT; ++t) {
      int n = p * NPASS + t * 16 + l16;
      float bY = by[n], bG = bg[n], uu = u[n];
      #pragma unroll
      for (int r = 0; r < 8; ++r) {
        float y = tanhf(accY[t][r] + bY);
        float g = 1.f / (1.f + expf(-(accG[t][r] + bG)));
        vsum[r] += y * g * uu;
      }
    }
  }
  // reduce over the 16 lanes of each half-wave (each lane covered disjoint N)
  #pragma unroll
  for (int r = 0; r < 8; ++r) {
    float s = vsum[r];
    s += __shfl_xor(s, 1, 32);
    s += __shfl_xor(s, 2, 32);
    s += __shfl_xor(s, 4, 32);
    s += __shfl_xor(s, 8, 32);
    vsum[r] = s;
  }
  int mRow = mBase + wave * 16;
  if (lane == 0) {
    #pragma unroll
    for (int r = 0; r < 8; ++r) vout[mRow + r] = vsum[r];
  } else if (lane == 16) {
    #pragma unroll
    for (int r = 0; r < 8; ++r) vout[mRow + 8 + r] = vsum[r];
  }
}

// ---------- kernel 3: logits + Gumbel + masked argmax -> one-hot ----------
__global__ void finalize_k(const float* __restrict__ vh, const float* __restrict__ vq,
                           const float* __restrict__ c0p, const float* __restrict__ noise,
                           const float* __restrict__ Wa, const float* __restrict__ ba,
                           float* __restrict__ out) {
  int tid = blockIdx.x * blockDim.x + threadIdx.x;
  if (tid >= MM) return;
  int i = tid % RR;
  int bBase = (tid / RR) * RR;
  float wa0 = Wa[0], wa1 = Wa[1], b0 = ba[0], c0 = *c0p;
  float qv = vq[tid];
  float best = -3e38f;
  int bestj = 0;
  for (int j = 0; j < RR; ++j) {
    float zj;
    if (j <= i) {
      float score = qv + vh[bBase + j] + c0;
      float dt = (float)(i - j + 1);            // cumsum(tril) closed form
      float logit = score * wa0 + dt * wa1 + b0;
      float nz = noise[tid * RR + j];
      float g = -logf(EPSF - logf(nz + EPSF));  // Gumbel(0,1)
      zj = logit + g;
    } else {
      zj = -1e9f;
    }
    if (zj > best) { best = zj; bestj = j; }    // first-max tie rule like jnp.argmax
  }
  for (int j = 0; j < RR; ++j) out[tid * RR + j] = (j == bestj) ? 1.f : 0.f;
}

extern "C" void kernel_launch(void* const* d_in, const int* in_sizes, int n_in,
                              void* d_out, int out_size, void* d_ws, size_t ws_size,
                              hipStream_t stream) {
  const float* hist  = (const float*)d_in[0];
  const float* ques  = (const float*)d_in[1];
  const float* noise = (const float*)d_in[2];
  const float* Wy_h  = (const float*)d_in[3];
  const float* by_h  = (const float*)d_in[4];
  const float* Wg_h  = (const float*)d_in[5];
  const float* bg_h  = (const float*)d_in[6];
  const float* Wy_q  = (const float*)d_in[7];
  const float* by_q  = (const float*)d_in[8];
  const float* Wg_q  = (const float*)d_in[9];
  const float* bg_q  = (const float*)d_in[10];
  const float* W1    = (const float*)d_in[11];
  const float* b1    = (const float*)d_in[12];
  const float* W2    = (const float*)d_in[13];
  const float* b2    = (const float*)d_in[14];
  const float* Wa    = (const float*)d_in[15];
  const float* ba    = (const float*)d_in[16];
  float* out = (float*)d_out;

  // ---- workspace layout ----
  char* ws = (char*)d_ws;
  _Float16* Xh16 = (_Float16*)ws;                        // [M][1024]
  _Float16* Xq16 = Xh16 + (size_t)MM * KK;               // [M][1024]
  _Float16* WytH = Xq16 + (size_t)MM * KK;               // [512][1024] x4
  _Float16* WgtH = WytH + (size_t)LLn * KK;
  _Float16* WytQ = WgtH + (size_t)LLn * KK;
  _Float16* WgtQ = WytQ + (size_t)LLn * KK;
  float* u  = (float*)(WgtQ + (size_t)LLn * KK);         // 1024
  float* c0 = u + KK;                                    // 1
  float* vh = u + KK + 16;                               // 10240
  float* vq = vh + MM;                                   // 10240

  const int n4 = (MM * KK) / 4;
  cvt_x<<<(n4 + 255) / 256, 256, 0, stream>>>(hist, Xh16, n4);
  cvt_x<<<(n4 + 255) / 256, 256, 0, stream>>>(ques, Xq16, n4);
  const int nw = LLn * KK;
  cvt_wt<<<(nw + 255) / 256, 256, 0, stream>>>(Wy_h, WytH);
  cvt_wt<<<(nw + 255) / 256, 256, 0, stream>>>(Wg_h, WgtH);
  cvt_wt<<<(nw + 255) / 256, 256, 0, stream>>>(Wy_q, WytQ);
  cvt_wt<<<(nw + 255) / 256, 256, 0, stream>>>(Wg_q, WgtQ);

  fold_mlp<<<(KK + 1 + 127) / 128, 128, 0, stream>>>(W1, b1, W2, b2, u, c0);

  gated_gemm<<<MM / MTILE, 128, 0, stream>>>(Xh16, WytH, WgtH, by_h, bg_h, u,       vh);
  gated_gemm<<<MM / MTILE, 128, 0, stream>>>(Xq16, WytQ, WgtQ, by_q, bg_q, u + LLn, vq);

  finalize_k<<<(MM + 255) / 256, 256, 0, stream>>>(vh, vq, c0, noise, Wa, ba, out);
}
